// AttentionBlock_71966472012478
// MI455X (gfx1250) — compile-verified, matrix operand order checked by reading, not run on hardware
//
#include <hip/hip_runtime.h>
#include <hip/hip_bf16.h>

// ---------------------------------------------------------------------------
// AttentionBlock for MI455X (gfx1250, wave32): bf16 WMMA, double-buffered
// TDM / async-to-LDS staging.  B=8 S=1024 E=1024 H=16 HD=64 FF=4096
// ---------------------------------------------------------------------------

typedef __attribute__((ext_vector_type(16))) __bf16 v16bf;
typedef __attribute__((ext_vector_type(8)))  __bf16 bf16x8;
typedef __attribute__((ext_vector_type(8)))  float  v8f;
typedef __attribute__((ext_vector_type(4)))  unsigned int u32x4;
typedef __attribute__((ext_vector_type(8)))  int i32x8;
typedef __attribute__((ext_vector_type(4)))  int i32x4;

#ifndef __has_builtin
#define __has_builtin(x) 0
#endif

#if __has_builtin(__builtin_amdgcn_tensor_load_to_lds) && \
    __has_builtin(__builtin_amdgcn_s_wait_tensorcnt)
#define HAS_TDM 1
#else
#define HAS_TDM 0
#endif

#if __has_builtin(__builtin_amdgcn_global_load_async_to_lds_b128)
#define HAS_ASYNC 1
#else
#define HAS_ASYNC 0
#endif

#if defined(__has_include)
#if __has_include(<hip/amd_detail/amd_gfx1250_TDM.h>)
#define TDM_6ARG 1
#else
#define TDM_6ARG 0
#endif
#else
#define TDM_6ARG 0
#endif

// ---------------------------------------------------------------------------

__device__ __forceinline__ v8f wmma_bf16(v16bf a, v16bf b, v8f c) {
    return __builtin_amdgcn_wmma_f32_16x16x32_bf16(
        false, a, false, b, (short)0, c, false, false);
}

// 16x32-bf16 A/B fragment from an LDS row: two contiguous 8-element runs.
__device__ __forceinline__ v16bf frag_from_lds(const __bf16* row, int half, int k0) {
    bf16x8 r0 = *(const bf16x8*)(row + k0 + half * 8);
    bf16x8 r1 = *(const bf16x8*)(row + k0 + 16 + half * 8);
    v16bf f;
#pragma unroll
    for (int i = 0; i < 8; ++i) { f[i] = r0[i]; f[i + 8] = r1[i]; }
    return f;
}

#if HAS_ASYNC
// Probe-discovered prototype: both pointers are plain `<4 x i32>*`.
__device__ __forceinline__ void async_cp16(const void* g, void* l) {
    __builtin_amdgcn_global_load_async_to_lds_b128((i32x4*)g, (i32x4*)l, 0, 0);
}
__device__ __forceinline__ void async_wait0() {
#if __has_builtin(__builtin_amdgcn_s_wait_asynccnt)
    __builtin_amdgcn_s_wait_asynccnt(0);
#else
    asm volatile("s_wait_asynccnt 0x0" ::: "memory");
#endif
}
#endif

#if HAS_TDM
// 2D bf16 tile load via Tensor Data Mover.
// LDS rows are padded 64B data + 16B pad (stride 40 elements): pad_interval
// code 3 (=16 dwords), pad_amount code 3 (=4 dwords).
__device__ __forceinline__ void tdm_load_tile_2d(const __bf16* gsrc,
                                                 uint32_t lds_byte_off,
                                                 uint32_t tensor_d0, uint32_t tensor_d1,
                                                 uint32_t tile_d0, uint32_t tile_d1,
                                                 uint64_t row_stride_elems) {
    uint64_t ga = (uint64_t)(uintptr_t)gsrc;
    u32x4 g0;
    g0[0] = 1u;                                  // count=1, user descriptor
    g0[1] = lds_byte_off;                        // lds_addr
    g0[2] = (uint32_t)ga;                        // global_addr[31:0]
    g0[3] = (uint32_t)((ga >> 32) & 0x01FFFFFFu) | (2u << 30);  // addr[56:32]|type=2
    i32x8 g1;
    g1[0] = (int)((1u << 16) | (1u << 20) | (3u << 22) | (3u << 25));
    //            data_size=2B  pad_en      pad_int=16dw  pad_amt=4dw
    g1[1] = (int)((tensor_d0 & 0xFFFFu) << 16);                       // dim0[15:0]
    g1[2] = (int)((tensor_d0 >> 16) | ((tensor_d1 & 0xFFFFu) << 16)); // dim0[31:16],dim1[15:0]
    g1[3] = (int)((tensor_d1 >> 16) | (tile_d0 << 16));               // dim1[31:16],tile0
    g1[4] = (int)(tile_d1 & 0xFFFFu);                                 // tile1, tile2=0
    g1[5] = (int)(uint32_t)(row_stride_elems & 0xFFFFFFFFu);          // dim0_stride lo
    g1[6] = (int)(uint32_t)((row_stride_elems >> 32) & 0xFFFFu);      // dim0_stride hi
    g1[7] = 0;
    i32x4 z4 = {0, 0, 0, 0};
#if TDM_6ARG
    i32x8 z8 = {0, 0, 0, 0, 0, 0, 0, 0};
    __builtin_amdgcn_tensor_load_to_lds(g0, g1, z4, z4, z8, 0);
#else
    __builtin_amdgcn_tensor_load_to_lds(g0, g1, z4, z4, 0);
#endif
}
#endif

// ---------------------------------------------------------------------------
// f32 -> bf16 conversion (one-shot for weights and x)
// ---------------------------------------------------------------------------
__global__ void __launch_bounds__(256)
f32_to_bf16(const float* __restrict__ s, __bf16* __restrict__ d, int n) {
    int i = (blockIdx.x * 256 + threadIdx.x) * 8;
    if (i + 8 <= n) {
        float4 a = ((const float4*)(s + i))[0];
        float4 b = ((const float4*)(s + i))[1];
        bf16x8 o;
        o[0] = (__bf16)a.x; o[1] = (__bf16)a.y; o[2] = (__bf16)a.z; o[3] = (__bf16)a.w;
        o[4] = (__bf16)b.x; o[5] = (__bf16)b.y; o[6] = (__bf16)b.z; o[7] = (__bf16)b.w;
        *(bf16x8*)(d + i) = o;
    }
}

// ---------------------------------------------------------------------------
// GEMM: C[M,N] = A[M,K] @ W[N,K]^T (A,W bf16; accumulate f32)
// EPI: 0 none | 1 relu(acc+bias) | 2 acc+bias+resid.  OUT_T: float or __bf16.
// Block 256 thr (8 waves), tile BM=128 BN=64 BK=32, wave = 32x32 C tile.
// Double-buffered staging: TDM (wave 0) > async-to-LDS > sync copy.
// One barrier per k-step; DMA of tile t+1 overlaps WMMA on tile t.
// ---------------------------------------------------------------------------
template <int EPI, typename OUT_T>
__global__ void __launch_bounds__(256)
gemm_bf16_wmma(const __bf16* __restrict__ A, const __bf16* __restrict__ W,
               const float* __restrict__ bias, const float* __restrict__ resid,
               OUT_T* __restrict__ C, int M, int N, int K) {
    __shared__ __align__(16) __bf16 sA[2][128][40];
    __shared__ __align__(16) __bf16 sW[2][64][40];

    const int tid  = threadIdx.x;
    const int lane = tid & 31;
    const int wave = tid >> 5;
    const int half = lane >> 4;
    const int l16  = lane & 15;
    const int wm   = wave & 3;
    const int wn   = wave >> 2;
    const int m0   = blockIdx.y * 128;
    const int n0   = blockIdx.x * 64;

    auto stage = [&](int buf, int k0) {
#if HAS_TDM
        if (tid < 32) {
            tdm_load_tile_2d(A + (size_t)m0 * K + k0,
                             (uint32_t)(uintptr_t)&sA[buf][0][0],
                             (uint32_t)K, (uint32_t)M, 32u, 128u, (uint64_t)K);
            tdm_load_tile_2d(W + (size_t)n0 * K + k0,
                             (uint32_t)(uintptr_t)&sW[buf][0][0],
                             (uint32_t)K, (uint32_t)N, 32u, 64u, (uint64_t)K);
        }
#elif HAS_ASYNC
#pragma unroll
        for (int i = 0; i < 2; ++i) {          // A: 512 16B chunks / 256 thr
            int c = tid * 2 + i;
            int row = c >> 2, cc = (c & 3) * 8;
            async_cp16(A + (size_t)(m0 + row) * K + k0 + cc, &sA[buf][row][cc]);
        }
        {
            int row = tid >> 2, cc = (tid & 3) * 8;   // W: 256 chunks / 256 thr
            async_cp16(W + (size_t)(n0 + row) * K + k0 + cc, &sW[buf][row][cc]);
        }
#else
        {
            int row = tid >> 1, cc = (tid & 1) * 16;
#pragma unroll
            for (int i = 0; i < 2; ++i)
                *(bf16x8*)&sA[buf][row][cc + i * 8] =
                    *(const bf16x8*)(A + (size_t)(m0 + row) * K + k0 + cc + i * 8);
            int wr = tid >> 2, wc = (tid & 3) * 8;
            *(bf16x8*)&sW[buf][wr][wc] =
                *(const bf16x8*)(W + (size_t)(n0 + wr) * K + k0 + wc);
            if (k0 + 32 < K) {
                __builtin_prefetch(A + (size_t)(m0 + row) * K + k0 + 32, 0, 0);
                __builtin_prefetch(W + (size_t)(n0 + wr) * K + k0 + 32, 0, 0);
            }
        }
#endif
    };
    auto stage_wait = [&]() {
#if HAS_TDM
        if (tid < 32) __builtin_amdgcn_s_wait_tensorcnt(0);
#elif HAS_ASYNC
        async_wait0();
#endif
    };

    v8f acc[2][2] = {};

    stage(0, 0);   // prologue
    for (int k0 = 0; k0 < K; k0 += 32) {
        const int cur = (k0 >> 5) & 1;
        stage_wait();          // tile issued last iteration has landed
        __syncthreads();       // everyone finished reading the other buffer
        if (k0 + 32 < K) stage(cur ^ 1, k0 + 32);

        v16bf af[2], wf[2];
#pragma unroll
        for (int mt = 0; mt < 2; ++mt)
            af[mt] = frag_from_lds(&sA[cur][wm * 32 + mt * 16 + l16][0], half, 0);
#pragma unroll
        for (int nt = 0; nt < 2; ++nt)
            wf[nt] = frag_from_lds(&sW[cur][wn * 32 + nt * 16 + l16][0], half, 0);

#pragma unroll
        for (int mt = 0; mt < 2; ++mt)
#pragma unroll
            for (int nt = 0; nt < 2; ++nt)
                acc[mt][nt] = wmma_bf16(af[mt], wf[nt], acc[mt][nt]);
    }

#pragma unroll
    for (int mt = 0; mt < 2; ++mt) {
#pragma unroll
        for (int nt = 0; nt < 2; ++nt) {
            const int n = n0 + wn * 32 + nt * 16 + l16;
            float bv = (EPI != 0) ? bias[n] : 0.0f;
#pragma unroll
            for (int r = 0; r < 8; ++r) {
                const int m = m0 + wm * 32 + mt * 16 + r + 8 * half;
                float v = acc[mt][nt][r];
                if (EPI == 1)      v = fmaxf(v + bv, 0.0f);
                else if (EPI == 2) v = v + bv + resid[(size_t)m * N + n];
                C[(size_t)m * N + n] = (OUT_T)v;
            }
        }
    }
}

// ---------------------------------------------------------------------------
// Flash attention over bf16 qkv [B,S,3E]; ctx out bf16 [B,S,E].
// Grid (S/64, H, B), 128 thr = 4 waves; wave owns 16 q rows.
// Double-buffered K/V staging (async K loads when available).
// ---------------------------------------------------------------------------
__global__ void __launch_bounds__(128)
attn_fa_wmma(const __bf16* __restrict__ qkv, __bf16* __restrict__ ctx) {
    __shared__ __align__(16) __bf16 sK[2][32][80];     // [key][d]
    __shared__ __align__(16) __bf16 sVT[2][64][40];    // [d][key]
    __shared__ __align__(16) __bf16 sP[4][16][40];     // per-wave P

    const int tid  = threadIdx.x;
    const int lane = tid & 31;
    const int w    = tid >> 5;
    const int half = lane >> 4;
    const int l16  = lane & 15;
    const int b  = blockIdx.z;
    const int hh = blockIdx.y;
    const int qt = blockIdx.x;

    const size_t RS = 3072;
    const __bf16* Qb = qkv + (size_t)b * 1024 * RS + hh * 64;
    const __bf16* Kb = Qb + 1024;
    const __bf16* Vb = Qb + 2048;

    const float SCALE = 1.7328679513998633f;   // sqrt(1/64)*2*ln(1024)
    const float L2E   = 1.4426950408889634f;

    // Q fragments, resident for the whole loop (scale is applied on scores)
    const int qrow = qt * 64 + w * 16 + l16;
    const __bf16* qp = Qb + (size_t)qrow * RS;
    v16bf qf[2];
#pragma unroll
    for (int c = 0; c < 2; ++c) {
        bf16x8 r0 = *(const bf16x8*)(qp + c * 32 + half * 8);
        bf16x8 r1 = *(const bf16x8*)(qp + c * 32 + 16 + half * 8);
#pragma unroll
        for (int i = 0; i < 8; ++i) { qf[c][i] = r0[i]; qf[c][i + 8] = r1[i]; }
    }

    auto stage_kv = [&](int buf, int kb) {
#if HAS_ASYNC
#pragma unroll
        for (int i = 0; i < 2; ++i) {               // K: 256 16B chunks / 128 thr
            int c = tid * 2 + i;
            int row = c >> 3, cc = (c & 7) * 8;
            async_cp16(Kb + (size_t)(kb * 32 + row) * RS + cc, &sK[buf][row][cc]);
        }
#else
        {
            int row = tid >> 2, cc = (tid & 3) * 16;
#pragma unroll
            for (int i = 0; i < 2; ++i)
                *(bf16x8*)&sK[buf][row][cc + i * 8] =
                    *(const bf16x8*)(Kb + (size_t)(kb * 32 + row) * RS + cc + i * 8);
        }
#endif
        // V transposed: needs per-element scatter, sync path
        {
            int row = tid >> 2, dl = (tid & 3) * 16;
            const __bf16* vp = Vb + (size_t)(kb * 32 + row) * RS + dl;
#pragma unroll
            for (int i = 0; i < 2; ++i) {
                bf16x8 t = *(const bf16x8*)(vp + i * 8);
#pragma unroll
                for (int j = 0; j < 8; ++j)
                    sVT[buf][dl + i * 8 + j][row] = t[j];
            }
        }
    };

    v8f accO[4] = {};
    float run_m[8], run_l[8];
#pragma unroll
    for (int r = 0; r < 8; ++r) { run_m[r] = -1e30f; run_l[r] = 0.0f; }

    stage_kv(0, 0);   // prologue
    for (int kb = 0; kb < 32; ++kb) {
        const int cur = kb & 1;
#if HAS_ASYNC
        async_wait0();
#endif
        __syncthreads();                 // kv[cur] landed; prior reads done
        if (kb + 1 < 32) stage_kv(cur ^ 1, kb + 1);

        // ---- S = Q K^T (two 16x16 tiles over 64-deep K) ----
        v8f s[2] = {};
#pragma unroll
        for (int nt = 0; nt < 2; ++nt) {
#pragma unroll
            for (int c = 0; c < 2; ++c) {
                v16bf kf = frag_from_lds(&sK[cur][nt * 16 + l16][0], half, c * 32);
                s[nt] = wmma_bf16(qf[c], kf, s[nt]);
            }
        }

        // ---- online softmax ----
        float p0[8], p1[8], fac[8];
#pragma unroll
        for (int r = 0; r < 8; ++r) {
            float s0 = s[0][r] * SCALE;
            float s1 = s[1][r] * SCALE;
            float mx = fmaxf(s0, s1);
#pragma unroll
            for (int msk = 1; msk < 16; msk <<= 1)
                mx = fmaxf(mx, __shfl_xor(mx, msk, 32));
            float nm = fmaxf(run_m[r], mx);
            fac[r]   = exp2f((run_m[r] - nm) * L2E);
            run_m[r] = nm;
            p0[r] = exp2f((s0 - nm) * L2E);
            p1[r] = exp2f((s1 - nm) * L2E);
            float sum = p0[r] + p1[r];
#pragma unroll
            for (int msk = 1; msk < 16; msk <<= 1)
                sum += __shfl_xor(sum, msk, 32);
            run_l[r] = run_l[r] * fac[r] + sum;
        }
#pragma unroll
        for (int g = 0; g < 4; ++g)
#pragma unroll
            for (int r = 0; r < 8; ++r)
                accO[g][r] = accO[g][r] * fac[r];

        // ---- P: C-layout -> A-fragment via LDS ----
#pragma unroll
        for (int r = 0; r < 8; ++r) {
            sP[w][r + 8 * half][l16]      = (__bf16)p0[r];
            sP[w][r + 8 * half][16 + l16] = (__bf16)p1[r];
        }
        __syncthreads();

        v16bf pf = frag_from_lds(&sP[w][l16][0], half, 0);
#pragma unroll
        for (int g = 0; g < 4; ++g) {
            v16bf vf = frag_from_lds(&sVT[cur][g * 16 + l16][0], half, 0);
            accO[g] = wmma_bf16(pf, vf, accO[g]);
        }
    }

#pragma unroll
    for (int g = 0; g < 4; ++g) {
#pragma unroll
        for (int r = 0; r < 8; ++r) {
            int m = qt * 64 + w * 16 + r + 8 * half;
            ctx[((size_t)(b * 1024 + m)) * 1024 + hh * 64 + g * 16 + l16] =
                (__bf16)(accO[g][r] / run_l[r]);
        }
    }
}

// ---------------------------------------------------------------------------
// out = LN(a (+ b)); optional bf16 shadow copy for downstream GEMM input.
// ---------------------------------------------------------------------------
__global__ void __launch_bounds__(256)
add_ln_kernel(const float* __restrict__ a, const float* __restrict__ b,
              const float* __restrict__ gamma, const float* __restrict__ beta,
              float* __restrict__ out, __bf16* __restrict__ outb) {
    __shared__ float red[16];
    const int row = blockIdx.x;
    const int tid = threadIdx.x;
    const int lane = tid & 31, wave = tid >> 5;

    float4 v = ((const float4*)(a + (size_t)row * 1024))[tid];
    if (b != nullptr) {
        float4 vb = ((const float4*)(b + (size_t)row * 1024))[tid];
        v.x += vb.x; v.y += vb.y; v.z += vb.z; v.w += vb.w;
    }
    float s = v.x + v.y + v.z + v.w;
    float q = v.x * v.x + v.y * v.y + v.z * v.z + v.w * v.w;
#pragma unroll
    for (int m = 1; m < 32; m <<= 1) {
        s += __shfl_xor(s, m, 32);
        q += __shfl_xor(q, m, 32);
    }
    if (lane == 0) { red[wave] = s; red[8 + wave] = q; }
    __syncthreads();
    if (tid == 0) {
        float S = 0.0f, Q = 0.0f;
#pragma unroll
        for (int i = 0; i < 8; ++i) { S += red[i]; Q += red[8 + i]; }
        red[0] = S; red[8] = Q;
    }
    __syncthreads();
    float mean = red[0] * (1.0f / 1024.0f);
    float var  = red[8] * (1.0f / 1024.0f) - mean * mean;
    float rs   = rsqrtf(var + 1e-5f);
    int c = tid * 4;
    float4 o;
    o.x = (v.x - mean) * rs * gamma[c + 0] + beta[c + 0];
    o.y = (v.y - mean) * rs * gamma[c + 1] + beta[c + 1];
    o.z = (v.z - mean) * rs * gamma[c + 2] + beta[c + 2];
    o.w = (v.w - mean) * rs * gamma[c + 3] + beta[c + 3];
    ((float4*)(out + (size_t)row * 1024))[tid] = o;
    if (outb != nullptr) {
        __bf16* p = outb + (size_t)row * 1024 + c;
        p[0] = (__bf16)o.x; p[1] = (__bf16)o.y; p[2] = (__bf16)o.z; p[3] = (__bf16)o.w;
    }
}

// ---------------------------------------------------------------------------
// Host orchestration
// ---------------------------------------------------------------------------
extern "C" void kernel_launch(void* const* d_in, const int* in_sizes, int n_in,
                              void* d_out, int out_size, void* d_ws, size_t ws_size,
                              hipStream_t stream) {
    (void)in_sizes; (void)n_in; (void)out_size; (void)ws_size;
    const float* x    = (const float*)d_in[0];
    const float* wqkv = (const float*)d_in[1];
    const float* wo   = (const float*)d_in[2];
    const float* g1   = (const float*)d_in[3];
    const float* be1  = (const float*)d_in[4];
    const float* g2   = (const float*)d_in[5];
    const float* be2  = (const float*)d_in[6];
    const float* w1   = (const float*)d_in[7];
    const float* bb1  = (const float*)d_in[8];
    const float* w2   = (const float*)d_in[9];
    const float* bb2  = (const float*)d_in[10];
    float* out = (float*)d_out;

    const int M = 8192, E = 1024, FF = 4096, E3 = 3072;

    // bump allocator over workspace (256B aligned)
    char* wsp = (char*)d_ws;
    auto alloc = [&](size_t bytes) {
        char* p = wsp;
        wsp += (bytes + 255) & ~(size_t)255;
        return p;
    };
    __bf16* wqkvb = (__bf16*)alloc((size_t)E3 * E * 2);   // 6 MiB
    __bf16* wob   = (__bf16*)alloc((size_t)E * E * 2);    // 2 MiB
    __bf16* w1b   = (__bf16*)alloc((size_t)FF * E * 2);   // 8 MiB
    __bf16* w2b   = (__bf16*)alloc((size_t)E * FF * 2);   // 8 MiB
    __bf16* xb    = (__bf16*)alloc((size_t)M * E * 2);    // 16 MiB
    __bf16* hb    = (__bf16*)alloc((size_t)M * E * 2);    // 16 MiB
    __bf16* ctxb  = (__bf16*)alloc((size_t)M * E * 2);    // 16 MiB
    float*  hf    = (float*)alloc((size_t)M * E * 4);     // 32 MiB
    float*  yf    = (float*)alloc((size_t)M * E * 4);     // 32 MiB
    __bf16* ffhb  = (__bf16*)alloc((size_t)M * FF * 2);   // 64 MiB
    __bf16* qkvb  = (__bf16*)alloc((size_t)M * E3 * 2);   // 48 MiB
    float*  attnf = (float*)qkvb;  // overlay: qkv dead once attention is done

    dim3 blk(256);

    // 0) one-shot bf16 conversions of x + all weights
    f32_to_bf16<<<dim3((M * E) / 2048), blk, 0, stream>>>(x, xb, M * E);
    f32_to_bf16<<<dim3((E3 * E) / 2048), blk, 0, stream>>>(wqkv, wqkvb, E3 * E);
    f32_to_bf16<<<dim3((E * E) / 2048), blk, 0, stream>>>(wo, wob, E * E);
    f32_to_bf16<<<dim3((FF * E) / 2048), blk, 0, stream>>>(w1, w1b, FF * E);
    f32_to_bf16<<<dim3((E * FF) / 2048), blk, 0, stream>>>(w2, w2b, E * FF);

    // 1) qkv = x @ Wqkv^T   (bf16 out)
    gemm_bf16_wmma<0, __bf16><<<dim3(E3 / 64, M / 128), blk, 0, stream>>>(
        xb, wqkvb, nullptr, nullptr, qkvb, M, E3, E);
    // 2) ctx = softmax(scale * Q K^T) V   (bf16 out)
    attn_fa_wmma<<<dim3(16, 16, 8), dim3(128), 0, stream>>>(qkvb, ctxb);
    // 3) attn = ctx @ Wo^T   (f32 out)
    gemm_bf16_wmma<0, float><<<dim3(E / 64, M / 128), blk, 0, stream>>>(
        ctxb, wob, nullptr, nullptr, attnf, M, E, E);
    // 4) h = LN1(x + attn)  (f32 + bf16 shadow)
    add_ln_kernel<<<dim3(M), blk, 0, stream>>>(x, attnf, g1, be1, hf, hb);
    // 5) ffh = relu(h @ W1^T + b1)  (bf16 out)
    gemm_bf16_wmma<1, __bf16><<<dim3(FF / 64, M / 128), blk, 0, stream>>>(
        hb, w1b, bb1, nullptr, ffhb, M, FF, E);
    // 6) y = ffh @ W2^T + b2 + h  (f32 out)
    gemm_bf16_wmma<2, float><<<dim3(E / 64, M / 128), blk, 0, stream>>>(
        ffhb, w2b, bb2, hf, yf, M, E, FF);
    // 7) out = LN2(y)
    add_ln_kernel<<<dim3(M), blk, 0, stream>>>(yf, nullptr, g2, be2, out, nullptr);
}
